// UNet3dDAttn_15135464751188
// MI455X (gfx1250) — matrix operand order repaired and checked by reading, not compiled, not run on hardware
//
#include <hip/hip_runtime.h>
#include <math.h>

// ---------------------------------------------------------------------------
// bf16 storage type + helpers (avoid scalar __bf16 arithmetic; only the WMMA
// fragment is typed as a __bf16 ext-vector).
// ---------------------------------------------------------------------------
typedef unsigned short bf16_t;
typedef __attribute__((ext_vector_type(16))) __bf16 v16bf;
typedef __attribute__((ext_vector_type(8)))  float  v8f;

__device__ __forceinline__ float bf2f(bf16_t h) {
  union { unsigned int u; float f; } c; c.u = ((unsigned int)h) << 16; return c.f;
}
__device__ __forceinline__ bf16_t f2bf(float f) {
  union { float f; unsigned int u; } c; c.f = f;
  unsigned int r = c.u + 0x7FFFu + ((c.u >> 16) & 1u);   // round-to-nearest-even
  return (bf16_t)(r >> 16);
}

// ---------------------------------------------------------------------------
// Weight repack: original (Co, Ci, 27) f32 -> bf16 fragments in the exact
// per-lane order this code feeds the WMMA A operand (B matches), with GEMM-K defined as
//   k = r*Ci + ci   (r = kd*9+kh*3+kw tap index, ci = input channel).
// Packed layout: wp[((coTile*nK + kt)*32 + lane)*16 + j], zero-padded K tail.
// A-layout (contiguous K): lane<16 -> row=lane, K {0..15};
// lane>=16 -> row=lane-16, K {16..31}.
// ---------------------------------------------------------------------------
__global__ void unet_pack_w(const float* __restrict__ w, bf16_t* __restrict__ wp,
                            int Ci, int Co, int nK) {
  int idx = blockIdx.x * blockDim.x + threadIdx.x;
  int total = (Co >> 4) * nK * 512;
  if (idx >= total) return;
  int j    = idx & 15;
  int lane = (idx >> 4) & 31;
  int tk   = idx >> 9;          // coTile*nK + kt
  int kt   = tk % nK;
  int coT  = tk / nK;
  int aRow = lane & 15;
  int aKb  = (lane < 16) ? 0 : 16;
  int kk = kt * 32 + aKb + j;
  int Ktot = Ci * 27;
  bf16_t v = 0;
  if (kk < Ktot) {
    int r = kk / Ci, ci = kk - r * Ci;
    v = f2bf(w[(size_t)(coT * 16 + aRow) * Ktot + (size_t)ci * 27 + r]);
  }
  wp[idx] = v;
}

// ---------------------------------------------------------------------------
// Fast implicit-GEMM conv (Ci power of two >= 32, Co multiple of 64; covers
// every layer but the first).  One wave -> 64(Co) x 16(spatial) f32 tile:
// per 32-wide K-chunk it loads ONE gathered B fragment (16 immediate-offset
// u16 loads + one boundary select) and FOUR pre-packed A fragments (8 x b128
// off one base register), then issues 4 back-to-back WMMAs on independent
// accumulators.  Tap index r is wave-uniform (SALU); no divergent branches.
// ---------------------------------------------------------------------------
__global__ __launch_bounds__(128) void unet_conv3d_wmma_f(
    const bf16_t* __restrict__ x, const bf16_t* __restrict__ wp,
    const float* __restrict__ bias, float* __restrict__ y,
    int lgCi, int nK, int D, int H, int W, int spTiles) {
  const int lane   = threadIdx.x & 31;
  const int wave   = threadIdx.x >> 5;
  const int spTile = blockIdx.x * 4 + wave;
  if (spTile >= spTiles) return;                 // wave-uniform
  const int HW = H * W;
  const int S  = D * HW;
  const int Ci = 1 << lgCi;
  const int co0 = blockIdx.y << 6;               // 4 co-tiles of 16 per wave

  const int nCol = lane & 15;                    // N column (B and C/D agree)
  const int sp   = spTile * 16 + nCol;           // S % 16 == 0 -> always valid
  int d0  = sp / HW;
  int rhw = sp - d0 * HW;
  int h0  = rhw / W;
  int w0  = rhw - h0 * W;
  const int bKb = (lane < 16) ? 0 : 16;          // B: lane half -> K offset

  const size_t tStr = (size_t)nK * 512;          // packed elems per co-tile
  const bf16_t* wptr = wp + (size_t)(blockIdx.y * 4) * tStr + lane * 16;

  v8f acc0 = {}, acc1 = {}, acc2 = {}, acc3 = {};
  for (int kt = 0; kt < nK; ++kt) {
    union { uint4 q[2]; v16bf v; } A0, A1, A2, A3;
    A0.q[0] = *(const uint4*)(wptr);
    A0.q[1] = *(const uint4*)(wptr + 8);
    A1.q[0] = *(const uint4*)(wptr + tStr);
    A1.q[1] = *(const uint4*)(wptr + tStr + 8);
    A2.q[0] = *(const uint4*)(wptr + 2 * tStr);
    A2.q[1] = *(const uint4*)(wptr + 2 * tStr + 8);
    A3.q[0] = *(const uint4*)(wptr + 3 * tStr);
    A3.q[1] = *(const uint4*)(wptr + 3 * tStr + 8);
    wptr += 512;

    int k0  = kt << 5;
    int r   = k0 >> lgCi;                        // uniform across chunk
    int ci0 = (k0 & (Ci - 1)) + bKb;
    int kd = r / 9, r9 = r - kd * 9;
    int kh = r9 / 3, kw = r9 - kh * 3;
    int id = d0 + kd - 1, ih = h0 + kh - 1, iw = w0 + kw - 1;
    bool ok = ((unsigned)id < (unsigned)D) & ((unsigned)ih < (unsigned)H) &
              ((unsigned)iw < (unsigned)W);
    int offsp = id * HW + ih * W + iw;           // compute unconditionally
    offsp = ok ? offsp : 0;                      // clamp -> valid address
    const bf16_t* bp = x + (size_t)ci0 * S + offsp;

    union { bf16_t u[16]; v16bf v; } B;
#pragma unroll
    for (int j = 0; j < 16; ++j) {
      bf16_t tv = bp[(size_t)j * S];             // unconditional load
      B.u[j] = ok ? tv : (bf16_t)0;              // v_cndmask, no branch
    }
    acc0 = __builtin_amdgcn_wmma_f32_16x16x32_bf16(
        false, A0.v, false, B.v, (short)0, acc0, false, false);
    acc1 = __builtin_amdgcn_wmma_f32_16x16x32_bf16(
        false, A1.v, false, B.v, (short)0, acc1, false, false);
    acc2 = __builtin_amdgcn_wmma_f32_16x16x32_bf16(
        false, A2.v, false, B.v, (short)0, acc2, false, false);
    acc3 = __builtin_amdgcn_wmma_f32_16x16x32_bf16(
        false, A3.v, false, B.v, (short)0, acc3, false, false);
  }
  // C/D layout: VGPR rI -> lanes 0-15: M=rI, lanes 16-31: M=rI+8; N=lane&15.
  const int mB = (lane < 16) ? 0 : 8;
#pragma unroll
  for (int rI = 0; rI < 8; ++rI) {
    int co = co0 + mB + rI;
    y[(size_t)co * S + sp]            = acc0[rI] + bias[co];
    y[(size_t)(co + 16) * S + sp]     = acc1[rI] + bias[co + 16];
    y[(size_t)(co + 32) * S + sp]     = acc2[rI] + bias[co + 32];
    y[(size_t)(co + 48) * S + sp]     = acc3[rI] + bias[co + 48];
  }
}

// ---------------------------------------------------------------------------
// Generic conv (used only for the first layer, Ci=3 / Ktot=81).  Same packed
// A; B gather is per-element but fully branchless (clamped addr + select).
// ---------------------------------------------------------------------------
__global__ __launch_bounds__(128) void unet_conv3d_wmma_g(
    const bf16_t* __restrict__ x, const bf16_t* __restrict__ wp,
    const float* __restrict__ bias, float* __restrict__ y,
    int Ci, int nK, int D, int H, int W, int spTiles) {
  const int lane   = threadIdx.x & 31;
  const int wave   = threadIdx.x >> 5;
  const int spTile = blockIdx.x * 4 + wave;
  if (spTile >= spTiles) return;
  const int HW = H * W;
  const int S  = D * HW;
  const int Ktot = Ci * 27;
  const int co0 = blockIdx.y << 4;

  const int nCol = lane & 15;
  const int sp   = spTile * 16 + nCol;
  int d0  = sp / HW;
  int rhw = sp - d0 * HW;
  int h0  = rhw / W;
  int w0  = rhw - h0 * W;
  const int bKb = (lane < 16) ? 0 : 16;

  const bf16_t* wptr = wp + (size_t)blockIdx.y * nK * 512 + lane * 16;

  v8f acc = {};
  for (int kt = 0; kt < nK; ++kt) {
    union { uint4 q[2]; v16bf v; } A;
    A.q[0] = *(const uint4*)(wptr);
    A.q[1] = *(const uint4*)(wptr + 8);
    wptr += 512;

    union { bf16_t u[16]; v16bf v; } B;
#pragma unroll
    for (int j = 0; j < 16; ++j) {
      int kk = kt * 32 + bKb + j;
      bool okk = kk < Ktot;
      int kkc = okk ? kk : 0;
      int r = kkc / Ci, ci = kkc - r * Ci;
      int kd = r / 9, r9 = r - kd * 9;
      int kh = r9 / 3, kw = r9 - kh * 3;
      int id = d0 + kd - 1, ih = h0 + kh - 1, iw = w0 + kw - 1;
      bool ok = okk & ((unsigned)id < (unsigned)D) &
                ((unsigned)ih < (unsigned)H) & ((unsigned)iw < (unsigned)W);
      size_t addr = ok ? ((size_t)(ci * D + id) * HW + ih * W + iw) : 0;
      bf16_t tv = x[addr];
      B.u[j] = ok ? tv : (bf16_t)0;
    }
    acc = __builtin_amdgcn_wmma_f32_16x16x32_bf16(
        false, A.v, false, B.v, (short)0, acc, false, false);
  }
  const int mB = (lane < 16) ? 0 : 8;
#pragma unroll
  for (int rI = 0; rI < 8; ++rI) {
    int co = co0 + mB + rI;
    y[(size_t)co * S + sp] = acc[rI] + bias[co];
  }
}

// ---------------------------------------------------------------------------
// Utility kernels
// ---------------------------------------------------------------------------
__global__ void unet_cvt_bf16(const float* __restrict__ s, bf16_t* __restrict__ d, int n) {
  int i = blockIdx.x * blockDim.x + threadIdx.x;
  if (i < n) d[i] = f2bf(s[i]);
}
__global__ void unet_copy_bf16(const bf16_t* __restrict__ s, bf16_t* __restrict__ d, int n) {
  int i = blockIdx.x * blockDim.x + threadIdx.x;
  if (i < n) d[i] = s[i];
}
__global__ void unet_zero_f32(float* __restrict__ p, int n) {
  int i = blockIdx.x * blockDim.x + threadIdx.x;
  if (i < n) p[i] = 0.0f;
}

// BN stats: per-channel sum / sumsq over S (grid = (C, chunks)).
__global__ void unet_bn_stats(const float* __restrict__ y, float* __restrict__ stats,
                              int C, int S) {
  int c = blockIdx.x;
  __shared__ float sS[256], sQ[256];
  float s = 0.f, q = 0.f;
  for (int i = blockIdx.y * blockDim.x + threadIdx.x; i < S;
       i += blockDim.x * gridDim.y) {
    float v = y[(size_t)c * S + i];
    s += v; q += v * v;
  }
  sS[threadIdx.x] = s; sQ[threadIdx.x] = q;
  __syncthreads();
  for (int o = 128; o > 0; o >>= 1) {
    if (threadIdx.x < o) {
      sS[threadIdx.x] += sS[threadIdx.x + o];
      sQ[threadIdx.x] += sQ[threadIdx.x + o];
    }
    __syncthreads();
  }
  if (threadIdx.x == 0) {
    atomicAdd(&stats[c], sS[0]);
    atomicAdd(&stats[C + c], sQ[0]);
  }
}

// BN (training-mode, biased var) + ReLU -> bf16.  grid = (ceil(S/256), C).
__global__ void unet_bn_relu(const float* __restrict__ y, const float* __restrict__ stats,
                             const float* __restrict__ g, const float* __restrict__ be,
                             bf16_t* __restrict__ out, int C, int S) {
  int i = blockIdx.x * blockDim.x + threadIdx.x;
  if (i >= S) return;
  int c = blockIdx.y;
  float m   = stats[c] / (float)S;
  float var = stats[C + c] / (float)S - m * m;
  float inv = rsqrtf(var + 1e-5f);
  float v = (y[(size_t)c * S + i] - m) * inv * g[c] + be[c];
  out[(size_t)c * S + i] = f2bf(fmaxf(v, 0.0f));
}

// MaxPool (1,2,2).  grid = (ceil(Ho*Wo/256), C, D).
__global__ void unet_maxpool122(const bf16_t* __restrict__ x, bf16_t* __restrict__ y,
                                int H, int W) {
  int Ho = H >> 1, Wo = W >> 1;
  int i = blockIdx.x * blockDim.x + threadIdx.x;
  if (i >= Ho * Wo) return;
  int c = blockIdx.y, d = blockIdx.z, D = gridDim.z;
  int ho = i / Wo, wo = i - ho * Wo;
  const bf16_t* p = x + ((size_t)c * D + d) * H * W + (2 * ho) * W + 2 * wo;
  float m = fmaxf(fmaxf(bf2f(p[0]), bf2f(p[1])), fmaxf(bf2f(p[W]), bf2f(p[W + 1])));
  y[((size_t)c * D + d) * Ho * Wo + i] = f2bf(m);
}

// d-attention: mean over (C,H,W) per depth slice.  grid = (D, chunksC).
__global__ void unet_dattn_mean(const bf16_t* __restrict__ x, float* __restrict__ dm,
                                int C, int D, int HW, float inv) {
  int d = blockIdx.x;
  __shared__ float sm[256];
  float s = 0.f;
  for (int c = blockIdx.y; c < C; c += gridDim.y) {
    const bf16_t* p = x + ((size_t)c * D + d) * HW;
    for (int i = threadIdx.x; i < HW; i += blockDim.x) s += bf2f(p[i]);
  }
  sm[threadIdx.x] = s;
  __syncthreads();
  for (int o = 128; o > 0; o >>= 1) {
    if (threadIdx.x < o) sm[threadIdx.x] += sm[threadIdx.x + o];
    __syncthreads();
  }
  if (threadIdx.x == 0) atomicAdd(&dm[d], sm[0] * inv);
}
// In-place x *= dm[d].  grid = (ceil(HW/256), C, D).
__global__ void unet_dattn_apply(bf16_t* __restrict__ x, const float* __restrict__ dm,
                                 int HW) {
  int i = blockIdx.x * blockDim.x + threadIdx.x;
  if (i >= HW) return;
  int c = blockIdx.y, d = blockIdx.z, D = gridDim.z;
  size_t idx = ((size_t)c * D + d) * HW + i;
  x[idx] = f2bf(bf2f(x[idx]) * dm[d]);
}

// ConvTranspose (1,2,2): non-overlapping 2x HW upsample.  tw: (Ci,Co,2,2) f32.
// grid = (ceil(Hc*Wc/256), Co, D).  Writes bf16 (into concat tail).
__global__ void unet_upconvT(const bf16_t* __restrict__ x, const float* __restrict__ tw,
                             const float* __restrict__ tb, bf16_t* __restrict__ y,
                             int Ci, int Co, int Hc, int Wc) {
  int i = blockIdx.x * blockDim.x + threadIdx.x;
  if (i >= Hc * Wc) return;
  int co = blockIdx.y, d = blockIdx.z, D = gridDim.z;
  int h = i / Wc, w = i - h * Wc;
  float a00 = tb[co], a01 = tb[co], a10 = tb[co], a11 = tb[co];
  for (int ci = 0; ci < Ci; ++ci) {
    float xv = bf2f(x[((size_t)ci * D + d) * Hc * Wc + i]);
    const float* wp = tw + ((size_t)ci * Co + co) * 4;
    a00 += xv * wp[0]; a01 += xv * wp[1]; a10 += xv * wp[2]; a11 += xv * wp[3];
  }
  int Hf = 2 * Hc, Wf = 2 * Wc;
  size_t base = ((size_t)co * D + d) * Hf * Wf + (size_t)(2 * h) * Wf + 2 * w;
  y[base]          = f2bf(a00);
  y[base + 1]      = f2bf(a01);
  y[base + Wf]     = f2bf(a10);
  y[base + Wf + 1] = f2bf(a11);
}

// Final 1x1 conv (C=64 -> 1) + sigmoid, f32 out.
__global__ void unet_outc(const bf16_t* __restrict__ h, const float* __restrict__ w,
                          const float* __restrict__ b, float* __restrict__ out,
                          int C, int S) {
  int i = blockIdx.x * blockDim.x + threadIdx.x;
  if (i >= S) return;
  float s = b[0];
  for (int c = 0; c < C; ++c) s += bf2f(h[(size_t)c * S + i]) * w[c];
  out[i] = 1.0f / (1.0f + expf(-s));
}

// ---------------------------------------------------------------------------
// Host-side orchestration helpers
// ---------------------------------------------------------------------------
static inline int cdiv(int a, int b) { return (a + b - 1) / b; }
static inline int ilog2(int v) { int l = 0; while ((1 << l) < v) ++l; return l; }

static void run_conv(hipStream_t s, const bf16_t* in, const float* wf32,
                     const float* bias, bf16_t* wbf, float* outf,
                     int Ci, int Co, int D, int H, int W) {
  int Ktot = Ci * 27;
  int nK = cdiv(Ktot, 32);
  int totalPack = (Co >> 4) * nK * 512;
  unet_pack_w<<<cdiv(totalPack, 256), 256, 0, s>>>(wf32, wbf, Ci, Co, nK);
  int S = D * H * W;
  int spTiles = S / 16;                   // S is always a multiple of 16 here
  if (Ci >= 32 && (Ci & (Ci - 1)) == 0 && (Co & 63) == 0) {
    dim3 g(cdiv(spTiles, 4), Co / 64);    // each wave: 4 co-tiles
    unet_conv3d_wmma_f<<<g, 128, 0, s>>>(in, wbf, bias, outf,
                                         ilog2(Ci), nK, D, H, W, spTiles);
  } else {
    dim3 g(cdiv(spTiles, 4), Co / 16);
    unet_conv3d_wmma_g<<<g, 128, 0, s>>>(in, wbf, bias, outf,
                                         Ci, nK, D, H, W, spTiles);
  }
}

static void run_bn_relu(hipStream_t s, const float* yf, const float* g,
                        const float* be, float* stats, bf16_t* out, int C, int S) {
  unet_zero_f32<<<cdiv(2 * C, 256), 256, 0, s>>>(stats, 2 * C);
  int chunks = S / 4096; if (chunks < 1) chunks = 1; if (chunks > 32) chunks = 32;
  dim3 g1(C, chunks);
  unet_bn_stats<<<g1, 256, 0, s>>>(yf, stats, C, S);
  dim3 g2(cdiv(S, 256), C);
  unet_bn_relu<<<g2, 256, 0, s>>>(yf, stats, g, be, out, C, S);
}

static void run_double_conv(hipStream_t s, const bf16_t* in, int Ci, int Co,
                            int D, int H, int W,
                            const float* w1, const float* b1, const float* g1, const float* be1,
                            const float* w2, const float* b2, const float* g2, const float* be2,
                            bf16_t* wbf, float* f32c, float* stats,
                            bf16_t* mid, bf16_t* out) {
  int S = D * H * W;
  run_conv(s, in, w1, b1, wbf, f32c, Ci, Co, D, H, W);
  run_bn_relu(s, f32c, g1, be1, stats, mid, Co, S);
  run_conv(s, mid, w2, b2, wbf, f32c, Co, Co, D, H, W);
  run_bn_relu(s, f32c, g2, be2, stats, out, Co, S);
}

static void run_dattn(hipStream_t s, bf16_t* x, float* dm, int C, int D, int HW) {
  unet_zero_f32<<<1, 32, 0, s>>>(dm, D);
  int cch = C < 16 ? C : 16;
  dim3 g1(D, cch);
  unet_dattn_mean<<<g1, 256, 0, s>>>(x, dm, C, D, HW, 1.0f / ((float)C * (float)HW));
  dim3 g2(cdiv(HW, 256), C, D);
  unet_dattn_apply<<<g2, 256, 0, s>>>(x, dm, HW);
}

static void run_up(hipStream_t s, const bf16_t* xc, const bf16_t* xs, int Ci,
                   int D, int Hc, int Wc,
                   const float* tw, const float* tb,
                   const float* w1, const float* b1, const float* g1, const float* be1,
                   const float* w2, const float* b2, const float* g2, const float* be2,
                   bf16_t* cat, bf16_t* wbf, float* f32c, float* stats,
                   bf16_t* mid, bf16_t* out) {
  int Co = Ci / 2;
  int Hf = 2 * Hc, Wf = 2 * Wc;
  int Sf = D * Hf * Wf;
  unet_copy_bf16<<<cdiv(Co * Sf, 256), 256, 0, s>>>(xs, cat, Co * Sf);
  dim3 g(cdiv(Hc * Wc, 256), Co, D);
  unet_upconvT<<<g, 256, 0, s>>>(xc, tw, tb, cat + (size_t)Co * Sf, Ci, Co, Hc, Wc);
  run_double_conv(s, cat, Ci, Co, D, Hf, Wf,
                  w1, b1, g1, be1, w2, b2, g2, be2, wbf, f32c, stats, mid, out);
}

// ---------------------------------------------------------------------------
// Entry point.  Input order (flattened setup_inputs dict):
//  0:x  1-8:inc  9-16:down1  17-24:down2  25-32:down3  33-40:down4
//  41-50:up1  51-60:up2  61-70:up3  71-80:up4  81:outc.w  82:outc.b
// ---------------------------------------------------------------------------
extern "C" void kernel_launch(void* const* d_in, const int* in_sizes, int n_in,
                              void* d_out, int out_size, void* d_ws, size_t ws_size,
                              hipStream_t stream) {
  (void)in_sizes; (void)n_in; (void)out_size; (void)ws_size;
  const float* X = (const float*)d_in[0];
#define PRM(i) ((const float*)d_in[(i)])

  const int D = 8;
  const int S0 = 8 * 128 * 128, S1 = 8 * 64 * 64, S2 = 8 * 32 * 32,
            S3 = 8 * 16 * 16;

  char* base = (char*)d_ws;
  size_t off = 0;
  auto alloc = [&](size_t bytes) -> void* {
    void* p = base + off;
    off += (bytes + 255) & ~(size_t)255;
    return p;
  };

  bf16_t* xin  = (bf16_t*)alloc((size_t)3   * S0 * 2);
  bf16_t* x1   = (bf16_t*)alloc((size_t)64  * S0 * 2);
  bf16_t* x2   = (bf16_t*)alloc((size_t)128 * S1 * 2);
  bf16_t* x3   = (bf16_t*)alloc((size_t)256 * S2 * 2);
  bf16_t* x4   = (bf16_t*)alloc((size_t)512 * S3 * 2);
  bf16_t* x5   = (bf16_t*)alloc((size_t)1024 * 512 * 2);
  bf16_t* wbf  = (bf16_t*)alloc((size_t)512 * 1024 * 27 * 2 + 4096); // packed W
  float*  f32c = (float*) alloc((size_t)64  * S0 * 4);
  bf16_t* mid  = (bf16_t*)alloc((size_t)64  * S0 * 2);
  bf16_t* cat  = (bf16_t*)alloc((size_t)128 * S0 * 2);
  bf16_t* hbuf = (bf16_t*)alloc((size_t)64  * S0 * 2);
  bf16_t* pool = (bf16_t*)alloc((size_t)64  * S1 * 2);
  float*  stats= (float*) alloc(2048 * 4);
  float*  dm   = (float*) alloc(64 * 4);

  // Encoder -----------------------------------------------------------------
  unet_cvt_bf16<<<cdiv(3 * S0, 256), 256, 0, stream>>>(X, xin, 3 * S0);
  run_double_conv(stream, xin, 3, 64, D, 128, 128,
                  PRM(1), PRM(2), PRM(3), PRM(4), PRM(5), PRM(6), PRM(7), PRM(8),
                  wbf, f32c, stats, mid, x1);

  { dim3 g(cdiv(64 * 64, 256), 64, D);
    unet_maxpool122<<<g, 256, 0, stream>>>(x1, pool, 128, 128); }
  run_double_conv(stream, pool, 64, 128, D, 64, 64,
                  PRM(9), PRM(10), PRM(11), PRM(12), PRM(13), PRM(14), PRM(15), PRM(16),
                  wbf, f32c, stats, mid, x2);
  run_dattn(stream, x2, dm, 128, D, 64 * 64);

  { dim3 g(cdiv(32 * 32, 256), 128, D);
    unet_maxpool122<<<g, 256, 0, stream>>>(x2, pool, 64, 64); }
  run_double_conv(stream, pool, 128, 256, D, 32, 32,
                  PRM(17), PRM(18), PRM(19), PRM(20), PRM(21), PRM(22), PRM(23), PRM(24),
                  wbf, f32c, stats, mid, x3);
  run_dattn(stream, x3, dm, 256, D, 32 * 32);

  { dim3 g(cdiv(16 * 16, 256), 256, D);
    unet_maxpool122<<<g, 256, 0, stream>>>(x3, pool, 32, 32); }
  run_double_conv(stream, pool, 256, 512, D, 16, 16,
                  PRM(25), PRM(26), PRM(27), PRM(28), PRM(29), PRM(30), PRM(31), PRM(32),
                  wbf, f32c, stats, mid, x4);
  run_dattn(stream, x4, dm, 512, D, 16 * 16);

  { dim3 g(cdiv(8 * 8, 256), 512, D);
    unet_maxpool122<<<g, 256, 0, stream>>>(x4, pool, 16, 16); }
  run_double_conv(stream, pool, 512, 1024, D, 8, 8,
                  PRM(33), PRM(34), PRM(35), PRM(36), PRM(37), PRM(38), PRM(39), PRM(40),
                  wbf, f32c, stats, mid, x5);
  run_dattn(stream, x5, dm, 1024, D, 8 * 8);

  // Decoder -----------------------------------------------------------------
  run_up(stream, x5, x4, 1024, D, 8, 8, PRM(41), PRM(42),
         PRM(43), PRM(44), PRM(45), PRM(46), PRM(47), PRM(48), PRM(49), PRM(50),
         cat, wbf, f32c, stats, mid, hbuf);
  run_dattn(stream, hbuf, dm, 512, D, 16 * 16);

  run_up(stream, hbuf, x3, 512, D, 16, 16, PRM(51), PRM(52),
         PRM(53), PRM(54), PRM(55), PRM(56), PRM(57), PRM(58), PRM(59), PRM(60),
         cat, wbf, f32c, stats, mid, hbuf);
  run_dattn(stream, hbuf, dm, 256, D, 32 * 32);

  run_up(stream, hbuf, x2, 256, D, 32, 32, PRM(61), PRM(62),
         PRM(63), PRM(64), PRM(65), PRM(66), PRM(67), PRM(68), PRM(69), PRM(70),
         cat, wbf, f32c, stats, mid, hbuf);
  run_dattn(stream, hbuf, dm, 128, D, 64 * 64);

  run_up(stream, hbuf, x1, 128, D, 64, 64, PRM(71), PRM(72),
         PRM(73), PRM(74), PRM(75), PRM(76), PRM(77), PRM(78), PRM(79), PRM(80),
         cat, wbf, f32c, stats, mid, hbuf);
  run_dattn(stream, hbuf, dm, 64, D, 128 * 128);

  // Output head -------------------------------------------------------------
  unet_outc<<<cdiv(S0, 256), 256, 0, stream>>>(hbuf, PRM(81), PRM(82),
                                               (float*)d_out, 64, S0);
#undef PRM
}
